// LlamaMLP_2869038154253
// MI455X (gfx1250) — compile-verified, matrix operand order checked by reading, not run on hardware
//
#include <hip/hip_runtime.h>

// ---------- vector types matching CDNA5 WMMA operand shapes ----------
typedef __attribute__((ext_vector_type(2)))  __bf16          v2bf;
typedef __attribute__((ext_vector_type(8)))  __bf16          v8bf;
typedef __attribute__((ext_vector_type(16))) __bf16          v16bf;
typedef __attribute__((ext_vector_type(32))) __bf16          v32bf;
typedef __attribute__((ext_vector_type(8)))  float           v8f;
typedef __attribute__((ext_vector_type(8)))  unsigned short  v8us;
typedef __attribute__((ext_vector_type(8)))  unsigned int    v8u;
typedef __attribute__((ext_vector_type(4)))  unsigned int    u32x4;
typedef __attribute__((ext_vector_type(8)))  int             i32x8;
typedef __attribute__((ext_vector_type(4)))  int             i32x4;

#define SHUF16(lo, hi) __builtin_shufflevector(lo, hi, 0,1,2,3,4,5,6,7,8,9,10,11,12,13,14,15)
#define SHUF32(lo, hi) __builtin_shufflevector(lo, hi, \
    0,1,2,3,4,5,6,7,8,9,10,11,12,13,14,15, \
    16,17,18,19,20,21,22,23,24,25,26,27,28,29,30,31)

__device__ __forceinline__ unsigned short f2bf(float f) {
  unsigned int u = __builtin_bit_cast(unsigned int, f);
  u += 0x7FFFu + ((u >> 16) & 1u);           // round-to-nearest-even
  return (unsigned short)(u >> 16);
}

// Packed fp32x2 -> bf16x2 (v_cvt_pk_bf16_f32 when available).
__device__ __forceinline__ unsigned int pack2bf(float a, float b) {
#if __has_builtin(__builtin_amdgcn_cvt_pk_bf16_f32)
  return __builtin_bit_cast(unsigned int, __builtin_amdgcn_cvt_pk_bf16_f32(a, b));
#else
  return (unsigned int)f2bf(a) | ((unsigned int)f2bf(b) << 16);
#endif
}

// ---------------------------------------------------------------------
// Tensor Data Mover: 2D tile load (global -> LDS), data_size = 2 bytes.
// Descriptor layout per CDNA5 ISA 8.3/8.4 (D# group0 128b, group1 256b).
// ---------------------------------------------------------------------
__device__ __forceinline__ void tdm_load_tile_2d_bf16(
    unsigned lds_addr, const void* gaddr,
    unsigned tile_d0, unsigned tile_d1,
    unsigned long long tensor_d0, unsigned long long tensor_d1,
    unsigned long long d0_stride /* elements */)
{
  const unsigned long long ga = (unsigned long long)gaddr;
  u32x4 g0;
  g0[0] = 1u;                                            // count=1 (valid), user mode
  g0[1] = lds_addr;                                      // LDS byte address
  g0[2] = (unsigned)(ga & 0xFFFFFFFFu);                  // global_addr[31:0]
  g0[3] = (unsigned)((ga >> 32) & 0x1FFFFFFu) | (2u << 30); // addr[56:32], type=2

  union { unsigned long long q[4]; i32x8 v; } g1;
  g1.q[0] = ((unsigned long long)1u << 16)               // data_size = 2 bytes
          | ((tensor_d0 & 0xFFFFull) << 48);             // tensor_dim0[15:0]
  g1.q[1] = ((tensor_d0 >> 16) & 0xFFFFull)              // tensor_dim0[31:16]
          | ((tensor_d1 & 0xFFFFFFFFull) << 16)          // tensor_dim1[31:0]
          | ((unsigned long long)(tile_d0 & 0xFFFFu) << 48); // tile_dim0
  g1.q[2] = (unsigned long long)(tile_d1 & 0xFFFFu)      // tile_dim1 (tile_dim2=0)
          | ((d0_stride & 0xFFFFFFFFull) << 32);         // dim0_stride[31:0]
  g1.q[3] = ((d0_stride >> 32) & 0xFFFFull);             // dim0_stride[47:32]

  const i32x4 z4 = (i32x4)0;
#if __has_include(<hip/amd_detail/amd_gfx1250_TDM.h>)
  const i32x8 z8 = (i32x8)0;                             // 6-arg toolchain
  __builtin_amdgcn_tensor_load_to_lds(g0, g1.v, z4, z4, z8, 0);
#else
  __builtin_amdgcn_tensor_load_to_lds(g0, g1.v, z4, z4, 0);   // 5-arg ROCm 7.2
#endif
}

// =====================================================================
// Kernel 1: h = silu(x @ Wg) * (x @ Wu), output bf16 [M, I]
// Block: 256 thr (8 waves). Tile M=64, N=64 (4x2 waves, each 16x32).
// K step 32. 4 wmma per wave per step.
// =====================================================================
__global__ __launch_bounds__(256) void gateup_kernel(
    const float* __restrict__ x, const float* __restrict__ wg,
    const float* __restrict__ wu, unsigned short* __restrict__ hout,
    int M, int H, int I)
{
  __shared__ __align__(16) unsigned short sA[64 * 32];  // x tile, [m][k]
  __shared__ __align__(16) unsigned short sG[64 * 32];  // Wg tile, [n][k]
  __shared__ __align__(16) unsigned short sU[64 * 32];  // Wu tile, [n][k]

  const int t    = threadIdx.x;
  const int lane = t & 31;
  const int w    = t >> 5;
  const int wm   = w & 3;        // wave M tile 0..3
  const int wn   = w >> 2;       // wave N pair 0..1 (covers 32 cols)
  const int half = lane >> 4;
  const int row  = lane & 15;
  const int m0   = blockIdx.y * 64;
  const int n0   = blockIdx.x * 64;

  const int arow = t >> 2, acol = (t & 3) * 8;        // x: 64x32, 8 f32/thread
  const int wkp  = (t & 15) * 2, wng = (t >> 4) * 4;  // W: k-pair + 4 n's per thread

  v8f accG0 = {}, accG1 = {}, accU0 = {}, accU1 = {};

  for (int k0 = 0; k0 < H; k0 += 32) {
    if (k0 + 32 < H) {   // -> global_prefetch_b8
      __builtin_prefetch(&x [(size_t)(m0 + arow) * H + k0 + 32 + acol], 0, 1);
      __builtin_prefetch(&wg[(size_t)(k0 + 32 + wkp) * I + n0 + wng], 0, 1);
      __builtin_prefetch(&wu[(size_t)(k0 + 32 + wkp) * I + n0 + wng], 0, 1);
    }
    // ---- stage x tile (fp32 -> bf16, packed pairs) ----
    {
      const float* src = &x[(size_t)(m0 + arow) * H + k0 + acol];
      unsigned int p[4];
#pragma unroll
      for (int i = 0; i < 4; ++i)
        p[i] = pack2bf(src[2 * i], src[2 * i + 1]);
      *(uint4*)&sA[arow * 32 + acol] = make_uint4(p[0], p[1], p[2], p[3]);
    }
    // ---- stage Wg / Wu tiles transposed ([n][k]); two K rows packed into
    //      one dword per n -> ds_store_b32 ----
    {
      const float* g0p = &wg[(size_t)(k0 + wkp) * I + n0 + wng];
      const float* g1p = g0p + I;
      const float* u0p = &wu[(size_t)(k0 + wkp) * I + n0 + wng];
      const float* u1p = u0p + I;
#pragma unroll
      for (int i = 0; i < 4; ++i) {
        *(unsigned int*)&sG[(wng + i) * 32 + wkp] = pack2bf(g0p[i], g1p[i]);
        *(unsigned int*)&sU[(wng + i) * 32 + wkp] = pack2bf(u0p[i], u1p[i]);
      }
    }
    __syncthreads();

    // ---- fragments (16-bit 16x32 layout): K = {half*8..+7, 16+half*8..+7}
    const v8bf a0 = *(const v8bf*)&sA[(wm * 16 + row) * 32 + half * 8];
    const v8bf a1 = *(const v8bf*)&sA[(wm * 16 + row) * 32 + 16 + half * 8];
    const v16bf a = SHUF16(a0, a1);

    const int nb = wn * 32;
#pragma unroll
    for (int s = 0; s < 2; ++s) {
      const int nr = (nb + s * 16 + row) * 32;
      const v8bf g0 = *(const v8bf*)&sG[nr + half * 8];
      const v8bf g1 = *(const v8bf*)&sG[nr + 16 + half * 8];
      const v16bf bg = SHUF16(g0, g1);
      const v8bf u0 = *(const v8bf*)&sU[nr + half * 8];
      const v8bf u1 = *(const v8bf*)&sU[nr + 16 + half * 8];
      const v16bf bu = SHUF16(u0, u1);
      if (s == 0) {
        accG0 = __builtin_amdgcn_wmma_f32_16x16x32_bf16(false, a, false, bg, (short)0, accG0, false, false);
        accU0 = __builtin_amdgcn_wmma_f32_16x16x32_bf16(false, a, false, bu, (short)0, accU0, false, false);
      } else {
        accG1 = __builtin_amdgcn_wmma_f32_16x16x32_bf16(false, a, false, bg, (short)0, accG1, false, false);
        accU1 = __builtin_amdgcn_wmma_f32_16x16x32_bf16(false, a, false, bu, (short)0, accU1, false, false);
      }
    }
    __syncthreads();
  }

  // ---- epilogue: SwiGLU (fast rcp, no IEEE div), write bf16 h ----
#pragma unroll
  for (int s = 0; s < 2; ++s) {
    const v8f gacc = s ? accG1 : accG0;
    const v8f uacc = s ? accU1 : accU0;
#pragma unroll
    for (int r = 0; r < 8; ++r) {
      const int m = m0 + wm * 16 + r + 8 * half;
      const int n = n0 + wn * 32 + s * 16 + row;
      const float g = gacc[r];
      const float sig = __builtin_amdgcn_rcpf(1.0f + __expf(-g));  // v_rcp_f32
      const float h = g * sig * uacc[r];
      hout[(size_t)m * I + n] = f2bf(h);
    }
  }
}

// =====================================================================
// Kernel 2: out = sparsify_2to4(h) @ Wd via V_SWMMAC_F32_16X16X64_BF16
// Block: 256 thr (8 waves). Tile M=64, N=64 (4x2 waves, each 16x32).
// K step 64. Double-buffered: TDM prefetches the next h tile while the
// current tile is consumed (one tensor op in flight during compute).
// =====================================================================
__global__ __launch_bounds__(256) void down_sparse_kernel(
    const unsigned short* __restrict__ hbuf, const float* __restrict__ wd,
    float* __restrict__ out, int M, int H, int I)
{
  __shared__ __align__(16) unsigned short sH[2][64 * 64];  // h tiles (TDM)
  __shared__ __align__(16) unsigned short sW[2][64 * 64];  // Wd tiles [n][k]

  const int t    = threadIdx.x;
  const int lane = t & 31;
  const int w    = t >> 5;
  const int wm   = w & 3;
  const int wn   = w >> 2;
  const int half = lane >> 4;
  const int row  = lane & 15;
  const int m0   = blockIdx.y * 64;
  const int n0   = blockIdx.x * 64;

  const int wkp = (t & 31) * 2, wng = (t >> 5) * 8;   // Wd: k-pair + 8 n's/thread

  // LDS byte address of sH[0] for the TDM descriptor (flat addr low 32 bits).
  const unsigned sH_lds = (unsigned)(unsigned long long)(&sH[0][0]);

  v8f acc0 = {}, acc1 = {};

  // ---- prologue: start DMA + stage Wd for tile 0 ----
  if (w == 0)
    tdm_load_tile_2d_bf16(sH_lds, &hbuf[(size_t)m0 * I], 64, 64,
                          (unsigned long long)I, (unsigned long long)M,
                          (unsigned long long)I);
  {
    const float* s0 = &wd[(size_t)wkp * H + n0 + wng];
    const float* s1 = s0 + H;
#pragma unroll
    for (int i = 0; i < 8; ++i)
      *(unsigned int*)&sW[0][(wng + i) * 64 + wkp] = pack2bf(s0[i], s1[i]);
  }

  int cur = 0;
  for (int k0 = 0; k0 < I; k0 += 64) {
    const int nxt = cur ^ 1;
    const bool has_next = (k0 + 64 < I);

    // ---- issue next tile's DMA + stage next Wd tile (prev barrier makes
    //      the alternate buffers safe to overwrite) ----
    if (has_next) {
      if (w == 0)
        tdm_load_tile_2d_bf16(sH_lds + (unsigned)nxt * 64 * 64 * 2,
                              &hbuf[(size_t)m0 * I + k0 + 64], 64, 64,
                              (unsigned long long)I, (unsigned long long)M,
                              (unsigned long long)I);
      __builtin_prefetch(&wd[(size_t)(k0 + 64 + wkp) * H + n0 + wng], 0, 1);
      const float* s0 = &wd[(size_t)(k0 + 64 + wkp) * H + n0 + wng];
      const float* s1 = s0 + H;
#pragma unroll
      for (int i = 0; i < 8; ++i)
        *(unsigned int*)&sW[nxt][(wng + i) * 64 + wkp] = pack2bf(s0[i], s1[i]);
      if (w == 0) __builtin_amdgcn_s_wait_tensorcnt(1);  // oldest (cur) done
    } else {
      if (w == 0) __builtin_amdgcn_s_wait_tensorcnt(0);
    }
    __syncthreads();

    // ---- per-lane 2:4 sparsify of this lane's 32 original K values ----
    const unsigned short* sHc = sH[cur];
    const int abase = (wm * 16 + row) * 64;
    const v8us q0 = *(const v8us*)&sHc[abase + half * 16];
    const v8us q1 = *(const v8us*)&sHc[abase + half * 16 + 8];
    const v8us q2 = *(const v8us*)&sHc[abase + 32 + half * 16];
    const v8us q3 = *(const v8us*)&sHc[abase + 32 + half * 16 + 8];
    unsigned short hv[32];
#pragma unroll
    for (int i = 0; i < 8; ++i) {
      hv[i] = q0[i]; hv[8 + i] = q1[i]; hv[16 + i] = q2[i]; hv[24 + i] = q3[i];
    }

    v8u packed = {};
    unsigned int codes = 0;
#pragma unroll
    for (int j = 0; j < 8; ++j) {          // 8 groups of 4, top-2 by |bf16|
      const unsigned short vv[4] = {hv[4*j], hv[4*j+1], hv[4*j+2], hv[4*j+3]};
      const unsigned short av[4] = {(unsigned short)(vv[0] & 0x7FFF),
                                    (unsigned short)(vv[1] & 0x7FFF),
                                    (unsigned short)(vv[2] & 0x7FFF),
                                    (unsigned short)(vv[3] & 0x7FFF)};
      int i0 = 0;
      if (av[1] > av[i0]) i0 = 1;
      if (av[2] > av[i0]) i0 = 2;
      if (av[3] > av[i0]) i0 = 3;
      int i1 = (i0 == 0) ? 1 : 0;
#pragma unroll
      for (int s = 0; s < 4; ++s)
        if (s != i0 && av[s] > av[i1]) i1 = s;
      const int lo = (i0 < i1) ? i0 : i1;
      const int hi = (i0 < i1) ? i1 : i0;   // ISA: idx0 < idx1
      packed[j] = (unsigned int)vv[lo] | ((unsigned int)vv[hi] << 16);
      codes |= ((unsigned int)lo << (4 * j)) | ((unsigned int)hi << (4 * j + 2));
    }
    const v16bf aS = __builtin_bit_cast(v16bf, packed);

    // ---- two dense B fragments (64x16 bf16), reuse sparse A ----
    const unsigned short* sWc = sW[cur];
#pragma unroll
    for (int s = 0; s < 2; ++s) {
      const int nbase = (wn * 32 + s * 16 + row) * 64;
      const v8bf b0 = *(const v8bf*)&sWc[nbase + half * 16];
      const v8bf b1 = *(const v8bf*)&sWc[nbase + half * 16 + 8];
      const v8bf b2 = *(const v8bf*)&sWc[nbase + 32 + half * 16];
      const v8bf b3 = *(const v8bf*)&sWc[nbase + 32 + half * 16 + 8];
      const v16bf blo = SHUF16(b0, b1);
      const v16bf bhi = SHUF16(b2, b3);
      const v32bf b   = SHUF32(blo, bhi);
      if (s == 0)
        acc0 = __builtin_amdgcn_swmmac_f32_16x16x64_bf16(false, aS, false, b, acc0, (int)codes, false, false);
      else
        acc1 = __builtin_amdgcn_swmmac_f32_16x16x64_bf16(false, aS, false, b, acc1, (int)codes, false, false);
    }
    __syncthreads();
    cur = nxt;
  }

  // ---- epilogue: fp32 output ----
#pragma unroll
  for (int s = 0; s < 2; ++s) {
    const v8f acc = s ? acc1 : acc0;
#pragma unroll
    for (int r = 0; r < 8; ++r) {
      const int m = m0 + wm * 16 + r + 8 * half;
      const int n = n0 + wn * 32 + s * 16 + row;
      out[(size_t)m * H + n] = acc[r];
    }
  }
}

// =====================================================================
extern "C" void kernel_launch(void* const* d_in, const int* in_sizes, int n_in,
                              void* d_out, int out_size, void* d_ws, size_t ws_size,
                              hipStream_t stream) {
  const float* x  = (const float*)d_in[0];
  const float* wg = (const float*)d_in[1];
  const float* wu = (const float*)d_in[2];
  const float* wd = (const float*)d_in[3];
  float* outp = (float*)d_out;
  unsigned short* hbuf = (unsigned short*)d_ws;   // bf16 intermediate [M, I]

  const int H = 2048;
  const int I = 8192;
  const int M = in_sizes[0] / H;                  // B*S = 8192

  dim3 blk(256);
  dim3 gA(I / 64, M / 64);
  gateup_kernel<<<gA, blk, 0, stream>>>(x, wg, wu, hbuf, M, H, I);

  dim3 gB(H / 64, M / 64);
  down_sparse_kernel<<<gB, blk, 0, stream>>>(hbuf, wd, outp, M, H, I);
}
